// VAE_7232724927203
// MI455X (gfx1250) — matrix operand order repaired and checked by reading, not compiled, not run on hardware
//
#include <hip/hip_runtime.h>

// ---------------- problem dims (match setup_inputs) ----------------
#define BB    128   // batch
#define DX    128   // obs dim
#define TT    512   // timesteps
#define DZ    16    // latent dim
#define DU    8     // input dim
#define DN    512   // RNN units
#define KPART 128   // particles
#define NTHR  256   // 8 waves of 32

#define DTc      0.1f
#define LOG2PI   1.8378770664093453f
#define LOGK     4.8520302639196171f   // ln(128)

typedef __attribute__((ext_vector_type(16))) __bf16 v16bf;
typedef __attribute__((ext_vector_type(8)))  float  v8f;
typedef __attribute__((ext_vector_type(2)))  float  v2f;

// ---------------- LDS layout (~118 KB, fits 320 KB WGP; 2 WG/WGP) ----------------
struct Smem {
  __bf16 nbf[DN][32];        // n (dN x dz), zero-padded to K=32   32 KB
  __bf16 mbf[DN][DZ];        // m (dN x dz), read transposed       16 KB
  __bf16 bobsT[DX][32];      // B_obs^T (dx x dz), padded           8 KB
  __bf16 qzbf[32][KPART];    // particle state bf16, padded rows    8 KB
  float  qz[DZ][KPART];      // particle state f32 (resample src)   8 KB
  float  qres[DZ][KPART];    // resampled particles                 8 KB
  float  prior[DZ][KPART];   // transition prior mean               8 KB
  float  kg0[DZ][DX];        // Kalman gain t=0                     8 KB
  float  kg[DZ][DX];         // Kalman gain t>0                     8 KB
  float  work[DZ][32];       // Gauss-Jordan workspace              2 KB
  float  oma0[DZ*DZ], oma[DZ*DZ], lch0[DZ*DZ], lch[DZ*DZ];      // 4 KB
  float  bias[DN];           // Wu@v + h                            2 KB
  float  xt[DX];
  float  obsb[DX];
  float  varx[DX];
  float  invstdx[DX];
  float  varz0[DZ], varz[DZ], invstdz0[DZ], invstdz[DZ];
  float  pm0[DZ], kgx[DZ];
  float  ut[DU];
  float  logw[KPART];
  float  cw[KPART];
  int    idxr[KPART];
  float  red[DZ][DZ];
  float  red2[NTHR];
  float  scal[8];            // 0 logstdx_sum 1 logstdz 2 logstdz0 3 logdet0 4 logdet
};
static_assert(sizeof(Smem) <= 160 * 1024, "LDS budget exceeded");

// ---------------- WMMA wrappers ----------------
__device__ inline v8f wmma_bf16(v16bf a, v16bf b, v8f c) {
  return __builtin_amdgcn_wmma_f32_16x16x32_bf16(false, a, false, b, (short)0, c,
                                                 false, false);
}
__device__ inline v8f wmma_f32k4(v2f a, v2f b, v8f c) {
  return __builtin_amdgcn_wmma_f32_16x16x4_f32(false, a, false, b, (short)0, c,
                                               false, false);
}

// A-fragment K index map for 16-bit A 16x32 (contiguous: K = 16h + j, matches B)
__device__ inline int amapK(int j, int h) {
  return 16 * h + j;
}
// A 16x32 bf16 fragment, row-major A[M][lda], tile at (row0, k=0..31)
__device__ inline v16bf load_a_frag(const __bf16* A, int lda, int row0, int lane) {
  int r = lane & 15, h = lane >> 4;
  v16bf f;
#pragma unroll
  for (int j = 0; j < 16; ++j) f[j] = A[(row0 + r) * lda + amapK(j, h)];
  return f;
}
// A 16x32 bf16 fragment from transposed storage: A(r,k) = Mt[(k0+k)*ld + r]
__device__ inline v16bf load_at_frag(const __bf16* Mt, int ld, int k0, int lane) {
  int r = lane & 15, h = lane >> 4;
  v16bf f;
#pragma unroll
  for (int j = 0; j < 16; ++j) f[j] = Mt[(k0 + amapK(j, h)) * ld + r];
  return f;
}
// B 32x16 bf16 fragment, row-major B[K][ldb], cols col0..+15 (lane j -> K=16h+j)
__device__ inline v16bf load_b_frag(const __bf16* B, int ldb, int col0, int lane) {
  int c = lane & 15, h = lane >> 4;
  v16bf f;
#pragma unroll
  for (int j = 0; j < 16; ++j) f[j] = B[(16 * h + j) * ldb + col0 + c];
  return f;
}

// ---------------- RNG / math helpers ----------------
__device__ inline unsigned pcg(unsigned x) {
  x = x * 747796405u + 2891336453u;
  x = ((x >> ((x >> 28) + 4u)) ^ x) * 277803737u;
  return (x >> 22) ^ x;
}
__device__ inline unsigned hash4(unsigned a, unsigned b, unsigned c, unsigned d) {
  return pcg(a ^ pcg(b ^ pcg(c ^ pcg(d))));
}
__device__ inline float u01(unsigned h) { return ((float)h + 0.5f) * 2.3283064e-10f; }
// one counter -> two independent gaussians (Box-Muller pair, sin+cos)
__device__ inline void gauss2(unsigned h1, unsigned h2, float* e0, float* e1) {
  float r = __fsqrt_rn(-2.0f * __logf(u01(h1)));
  float sv, cv;
  __sincosf(6.28318530717958647f * u01(h2), &sv, &cv);
  *e0 = r * cv;
  *e1 = r * sv;
}
__device__ inline float fast_tanh(float x) {
  x = fminf(10.f, fmaxf(-10.f, x));
  float e = __expf(2.0f * x);
  return (e - 1.0f) / (e + 1.0f);
}

// ---------------- on-device 16x16 Kalman (inverse + cholesky) ----------------
__device__ void kalman16(Smem& S, int tid, const float* varz, const float* Bg,
                         float* kg, float* oma, float* lch, float* logdet_slot) {
  int r = tid >> 4, c = tid & 15;
  // prec = diag(1/varz) + B (1/varx) B^T ; augmented [prec | I]
  {
    float acc = (r == c) ? 1.0f / varz[r] : 0.0f;
    for (int d = 0; d < DX; ++d)
      acc += Bg[r * DX + d] * Bg[c * DX + d] / S.varx[d];
    S.work[r][c] = acc;
    S.work[r][16 + c] = (r == c) ? 1.0f : 0.0f;
  }
  // Gauss-Jordan
  for (int p = 0; p < DZ; ++p) {
    __syncthreads();
    if (tid < 32) {
      float piv = S.work[p][p];
      S.work[p][tid] = S.work[p][tid] / piv;  // each thread owns one column
    }
    __syncthreads();
    float f = (r != p) ? S.work[r][p] : 0.0f;
    __syncthreads();
    if (r != p) {
      S.work[r][c]      -= f * S.work[p][c];
      S.work[r][16 + c] -= f * S.work[p][16 + c];
    }
  }
  __syncthreads();
  // var_Q = 1e-8 I + 0.5 (inv + inv^T) -> work[:, 0:16]
  {
    float v = 0.5f * (S.work[r][16 + c] + S.work[c][16 + r]);
    if (r == c) v += 1e-8f;
    __syncthreads();
    S.work[r][c] = v;
  }
  __syncthreads();
  // Cholesky (right-looking) into lch
  lch[r * DZ + c] = (c <= r) ? S.work[r][c] : 0.0f;
  for (int k = 0; k < DZ; ++k) {
    __syncthreads();
    if (tid == 0) lch[k * DZ + k] = sqrtf(lch[k * DZ + k]);
    __syncthreads();
    if (tid > k && tid < DZ) lch[tid * DZ + k] /= lch[k * DZ + k];
    __syncthreads();
    if (c > k && c <= r) lch[r * DZ + c] -= lch[r * DZ + k] * lch[c * DZ + k];
  }
  __syncthreads();
  if (tid == 0) {
    float ld = 0.f;
    for (int z = 0; z < DZ; ++z) ld += __logf(lch[z * DZ + z]);
    *logdet_slot = ld;
  }
  // Kg = var_Q @ (B / varx)   (16 x 128)
  for (int e = tid; e < DZ * DX; e += NTHR) {
    int i = e >> 7, d = e & 127;
    float acc = 0.f;
#pragma unroll
    for (int s = 0; s < DZ; ++s) acc += S.work[i][s] * Bg[s * DX + d];
    kg[i * DX + d] = acc / S.varx[d];
  }
  __syncthreads();
  // oma = I - Kg @ B^T
  {
    float acc = 0.f;
    for (int d = 0; d < DX; ++d) acc += kg[r * DX + d] * Bg[c * DX + d];
    oma[r * DZ + c] = ((r == c) ? 1.0f : 0.0f) - acc;
  }
  __syncthreads();
}

// ---------------- systematic resampling ----------------
__device__ void resample(Smem& S, int b, int t, int tid) {
  __syncthreads();
  float lw = (tid < KPART) ? S.logw[tid] : -1e30f;
  S.red2[tid] = lw;
  __syncthreads();
  for (int off = 128; off >= 1; off >>= 1) {
    if (tid < off) S.red2[tid] = fmaxf(S.red2[tid], S.red2[tid + off]);
    __syncthreads();
  }
  float mx = S.red2[0];
  __syncthreads();
  if (tid < KPART) S.cw[tid] = __expf(lw - mx);
  __syncthreads();
  for (int off = 1; off < KPART; off <<= 1) {     // Hillis-Steele inclusive scan
    float v = (tid < KPART && tid >= off) ? S.cw[tid - off] : 0.f;
    __syncthreads();
    if (tid < KPART) S.cw[tid] += v;
    __syncthreads();
  }
  float tot = S.cw[KPART - 1];
  float u0 = u01(hash4(0xC0FFEEu, (unsigned)b, (unsigned)t, 0x51D0u));
  if (tid < KPART) {
    float pos = (u0 + (float)tid) * (1.0f / KPART) * tot;  // unnormalized compare
    int lo = 0, hi = KPART;
    while (lo < hi) {
      int mid = (lo + hi) >> 1;
      if (S.cw[mid] < pos) lo = mid + 1; else hi = mid;
    }
    S.idxr[tid] = (lo < KPART) ? lo : (KPART - 1);
  }
  __syncthreads();
  for (int e = tid; e < DZ * KPART; e += NTHR) {
    int z = e >> 7, k = e & 127;
    float v = S.qz[z][S.idxr[k]];
    S.qres[z][k] = v;
    S.qzbf[z][k] = (__bf16)v;
  }
  __syncthreads();
}

// ---------------- transition: prior = (1-dt) z + dt m^T tanh(n z + bias) ----------------
// fully fused: pre tiles (WMMA bf16) -> tanh in regs -> D->B frag via shfl_xor(16)
// -> accumulate m^T contraction (WMMA bf16). No big staging buffer.
__device__ void transition(Smem& S, int lane, int wid) {
  const int col0 = wid * 16, c = lane & 15, h = lane >> 4;
  v16bf bq = load_b_frag(&S.qzbf[0][0], KPART, col0, lane);
  v8f outc = {0.f, 0.f, 0.f, 0.f, 0.f, 0.f, 0.f, 0.f};
  v8f zero = {0.f, 0.f, 0.f, 0.f, 0.f, 0.f, 0.f, 0.f};
  for (int rt2 = 0; rt2 < DN / 32; ++rt2) {
    int r0 = rt2 * 32;
    v16bf a0 = load_a_frag(&S.nbf[0][0], 32, r0, lane);
    v8f p0 = wmma_bf16(a0, bq, zero);
    v16bf a1 = load_a_frag(&S.nbf[0][0], 32, r0 + 16, lane);
    v8f p1 = wmma_bf16(a1, bq, zero);
    float t0v[8], t1v[8];
#pragma unroll
    for (int v = 0; v < 8; ++v) {
      t0v[v] = fast_tanh(p0[v] + S.bias[r0 + v + 8 * h]);
      t1v[v] = fast_tanh(p1[v] + S.bias[r0 + 16 + v + 8 * h]);
    }
    // D-frag (row = v+8h) -> B-frag (lane j covers K = r0+16h+j): swap halves
    v16bf bfrag;
#pragma unroll
    for (int v = 0; v < 8; ++v) {
      float ex0 = __shfl_xor(t0v[v], 16);
      float ex1 = __shfl_xor(t1v[v], 16);
      bfrag[v]     = (__bf16)((h == 0) ? t0v[v] : ex1);
      bfrag[v + 8] = (__bf16)((h == 0) ? ex0 : t1v[v]);
    }
    v16bf am = load_at_frag(&S.mbf[0][0], DZ, r0, lane);  // A = m^T
    outc = wmma_bf16(am, bfrag, outc);
  }
#pragma unroll
  for (int v = 0; v < 8; ++v) {
    int row = v + 8 * h;
    S.prior[row][col0 + c] = (1.0f - DTc) * S.qres[row][col0 + c] + DTc * outc[v];
  }
}

// ---------------- proposal + log-weights + logsumexp ----------------
__device__ void proposal_obs_weights(Smem& S, int b, int t, int tid, int lane, int wid,
                                     const float* kg, const float* oma,
                                     const float* lch, const float* invstdz,
                                     float logstdz_sum, float logdet, float* elbo) {
  const int col0 = wid * 16, c = lane & 15, h = lane >> 4;
  __syncthreads();
  // kgx = Kg @ (xt - obs_bias)   (shared across particles)
  {
    int i = tid >> 4, part = tid & 15;
    float acc = 0.f;
#pragma unroll
    for (int q = 0; q < 8; ++q) {
      int d = part * 8 + q;
      acc += kg[i * DX + d] * (S.xt[d] - S.obsb[d]);
    }
    S.red[i][part] = acc;
  }
  __syncthreads();
  if (tid < DZ) {
    float a = 0.f;
    for (int p = 0; p < DZ; ++p) a += S.red[tid][p];
    S.kgx[tid] = a;
  }
  __syncthreads();

  // mean_Q = oma @ prior  (f32 WMMA, 4 chained K=4 steps, full precision)
  v8f mq = {0.f, 0.f, 0.f, 0.f, 0.f, 0.f, 0.f, 0.f};
  const int r = lane & 15;
#pragma unroll
  for (int kk = 0; kk < 4; ++kk) {
    v2f a2, b2;
#pragma unroll
    for (int v = 0; v < 2; ++v) {
      int k = kk * 4 + 2 * h + v;
      a2[v] = oma[r * DZ + k];
      b2[v] = S.prior[k][col0 + c];
    }
    mq = wmma_f32k4(a2, b2, mq);
  }

  // noise = L @ eps ; ll_qz quad = sum eps^2 (since Qz - mean_Q = L eps exactly)
  // one cheap counter-hash pair -> two gaussians per K=4 chunk (Box-Muller sin+cos)
  v8f ne = {0.f, 0.f, 0.f, 0.f, 0.f, 0.f, 0.f, 0.f};
  float esq = 0.f;
  unsigned sb = (unsigned)b * 1315423911u ^ (unsigned)t * 2654435761u;
#pragma unroll
  for (int kk = 0; kk < 4; ++kk) {
    int s0 = kk * 4 + 2 * h;   // this lane's first latent index in the chunk
    unsigned h1 = pcg(sb + (unsigned)s0 * 0x632BE59Bu +
                      (unsigned)(col0 + c) * 0x85EBCA77u);
    unsigned h2 = pcg(h1 ^ 0x9E3779B9u);
    float e0, e1;
    gauss2(h1, h2, &e0, &e1);
    esq += e0 * e0 + e1 * e1;
    v2f a2, e2;
    a2[0] = lch[r * DZ + s0];
    a2[1] = lch[r * DZ + s0 + 1];
    e2[0] = e0;
    e2[1] = e1;
    ne = wmma_f32k4(a2, e2, ne);
  }
  esq += __shfl_xor(esq, 16);  // pair covers all 16 latent dims

  // Qz_new = mean_Q + kgx + L eps ; ll_pz partial
  float pz = 0.f;
#pragma unroll
  for (int v = 0; v < 8; ++v) {
    int row = v + 8 * h;
    float q = mq[v] + S.kgx[row] + ne[v];
    float d = (q - S.prior[row][col0 + c]) * invstdz[row];
    pz += d * d;
    S.qz[row][col0 + c] = q;
    S.qzbf[row][col0 + c] = (__bf16)q;
  }
  pz += __shfl_xor(pz, 16);
  __syncthreads();

  // obs ll: mean_x = B_obs^T @ Qz_new (bf16 WMMA), rx accumulation
  float xs = 0.f;
  v8f zero = {0.f, 0.f, 0.f, 0.f, 0.f, 0.f, 0.f, 0.f};
  v16bf bq = load_b_frag(&S.qzbf[0][0], KPART, col0, lane);
#pragma unroll
  for (int rt = 0; rt < DX / 16; ++rt) {
    v16bf a = load_a_frag(&S.bobsT[0][0], 32, rt * 16, lane);
    v8f mx = wmma_bf16(a, bq, zero);
#pragma unroll
    for (int v = 0; v < 8; ++v) {
      int d = rt * 16 + v + 8 * h;
      float rx = (S.xt[d] - mx[v] - S.obsb[d]) * S.invstdx[d];
      xs += rx * rx;
    }
  }
  xs += __shfl_xor(xs, 16);

  float llx  = -0.5f * xs  - S.scal[0]   - 0.5f * DX * LOG2PI;
  float llpz = -0.5f * pz  - logstdz_sum - 0.5f * DZ * LOG2PI;
  float llqz = -0.5f * esq - logdet      - 0.5f * DZ * LOG2PI;
  if (h == 0) S.logw[col0 + c] = llx + llpz - llqz;
  __syncthreads();

  // logsumexp over K
  float val = (tid < KPART) ? S.logw[tid] : -1e30f;
  S.red2[tid] = val;
  __syncthreads();
  for (int off = 128; off >= 1; off >>= 1) {
    if (tid < off) S.red2[tid] = fmaxf(S.red2[tid], S.red2[tid + off]);
    __syncthreads();
  }
  float mm = S.red2[0];
  __syncthreads();
  S.red2[tid] = (tid < KPART) ? __expf(val - mm) : 0.f;
  __syncthreads();
  for (int off = 128; off >= 1; off >>= 1) {
    if (tid < off) S.red2[tid] += S.red2[tid + off];
    __syncthreads();
  }
  *elbo += mm + __logf(S.red2[0]) - LOGK;
  __syncthreads();
}

// ---------------- layout prep: (B,dx,T) -> (B,T,dx) for coalesced per-step loads ----
__global__ void transpose_bdt(const float* __restrict__ src, float* __restrict__ dst,
                              int D, long long total) {
  long long i = (long long)blockIdx.x * NTHR + threadIdx.x;
  if (i >= total) return;
  int d = (int)(i % D);
  int t = (int)((i / D) % TT);
  int b = (int)(i / ((long long)D * TT));
  dst[i] = src[((long long)b * D + d) * TT + t];
}

// ---------------- main particle-filter kernel: one workgroup per trial ----------------
__global__ __launch_bounds__(NTHR)
void pf_elbo_kernel(const float* __restrict__ xbuf, const float* __restrict__ ubuf,
                    int tr_layout,
                    const float* __restrict__ Rz, const float* __restrict__ Rz0,
                    const float* __restrict__ Rx, const float* __restrict__ Bobs,
                    const float* __restrict__ Obias, const float* __restrict__ Mw,
                    const float* __restrict__ Nw, const float* __restrict__ Wug,
                    const float* __restrict__ Hb, const float* __restrict__ W0g,
                    const float* __restrict__ b0g, float* __restrict__ out) {
  extern __shared__ __align__(16) char smem_raw[];
  Smem& S = *reinterpret_cast<Smem*>(smem_raw);
  const int tid = threadIdx.x, lane = tid & 31, wid = tid >> 5;
  const int b = blockIdx.x;

  // ---- stage weights to LDS (bf16, zero-padded to K=32 for WMMA) ----
  for (int e = tid; e < DN * 32; e += NTHR) {
    int j = e >> 5, k = e & 31;
    S.nbf[j][k] = (k < DZ) ? (__bf16)Nw[j * DZ + k] : (__bf16)0.f;
  }
  for (int e = tid; e < DN * DZ; e += NTHR) S.mbf[0][e] = (__bf16)Mw[e];
  for (int e = tid; e < DX * 32; e += NTHR) {
    int d = e >> 5, k = e & 31;
    S.bobsT[d][k] = (k < DZ) ? (__bf16)Bobs[k * DX + d] : (__bf16)0.f;
  }
  for (int e = tid; e < 32 * KPART; e += NTHR) S.qzbf[0][e] = (__bf16)0.f;
  if (tid < DX) {
    float rx = Rx[tid];
    float vx = fmaxf(rx * rx, 1e-8f);
    float sx = fmaxf(fabsf(rx), 1e-4f);
    S.varx[tid] = vx;
    S.invstdx[tid] = 1.0f / sx;
    S.obsb[tid] = Obias[tid];
  }
  if (tid < DZ) {
    float v = Rz[tid] * Rz[tid] + 1e-8f;
    float v0 = Rz0[tid] * Rz0[tid] + 1e-8f;
    S.varz[tid] = v;    S.invstdz[tid]  = __frsqrt_rn(v);
    S.varz0[tid] = v0;  S.invstdz0[tid] = __frsqrt_rn(v0);
  }
  __syncthreads();
  if (tid == 0) {
    float a = 0.f;
    for (int d = 0; d < DX; ++d) a += __logf(fmaxf(fabsf(Rx[d]), 1e-4f));
    S.scal[0] = a;
    float z = 0.f, z0 = 0.f;
    for (int q = 0; q < DZ; ++q) {
      z  += 0.5f * __logf(S.varz[q]);
      z0 += 0.5f * __logf(S.varz0[q]);
    }
    S.scal[1] = z;
    S.scal[2] = z0;
  }
  __syncthreads();

  // ---- on-device Kalman proposals (t=0 and t>0 variants) ----
  kalman16(S, tid, S.varz0, Bobs, &S.kg0[0][0], S.oma0, S.lch0, &S.scal[3]);
  kalman16(S, tid, S.varz,  Bobs, &S.kg[0][0],  S.oma,  S.lch,  &S.scal[4]);
  __syncthreads();

  const float logstdz = S.scal[1], logstdz0 = S.scal[2];
  const float logdet0 = S.scal[3], logdet = S.scal[4];
  float elbo = 0.f;

  // ---- t = 0 : prior from stimulus, Kalman proposal ----
  if (tid < DX) {
    S.xt[tid] = tr_layout ? xbuf[((long long)b * TT + 0) * DX + tid]
                          : xbuf[((long long)b * DX + tid) * TT + 0];
  }
  if (tid < DU) {
    S.ut[tid] = tr_layout ? ubuf[((long long)b * TT + 0) * DU + tid]
                          : ubuf[((long long)b * DU + tid) * TT + 0];
  }
  __syncthreads();
  if (tid < DZ) {
    float a = b0g[tid];
#pragma unroll
    for (int q = 0; q < DU; ++q) a += W0g[tid * DU + q] * S.ut[q];
    S.pm0[tid] = a;
  }
  __syncthreads();
  for (int e = tid; e < DZ * KPART; e += NTHR) S.prior[0][e] = S.pm0[e >> 7];
  __syncthreads();
  proposal_obs_weights(S, b, 0, tid, lane, wid, &S.kg0[0][0], S.oma0, S.lch0,
                       S.invstdz0, logstdz0, logdet0, &elbo);

  // ---- t = 1 .. T-1 ----
  for (int t = 1; t < TT; ++t) {
    resample(S, b, t, tid);
    if (tid < DX) {
      S.xt[tid] = tr_layout ? xbuf[((long long)b * TT + t) * DX + tid]
                            : xbuf[((long long)b * DX + tid) * TT + t];
    }
    if (tid < DU) {
      S.ut[tid] = tr_layout ? ubuf[((long long)b * TT + (t - 1)) * DU + tid]
                            : ubuf[((long long)b * DU + tid) * TT + (t - 1)];
    }
    __syncthreads();
    for (int e = tid; e < DN; e += NTHR) {  // bias = Wu@v + h
      float a = Hb[e];
#pragma unroll
      for (int q = 0; q < DU; ++q) a += Wug[e * DU + q] * S.ut[q];
      S.bias[e] = a;
    }
    __syncthreads();
    transition(S, lane, wid);
    __syncthreads();
    proposal_obs_weights(S, b, t, tid, lane, wid, &S.kg[0][0], S.oma, S.lch,
                         S.invstdz, logstdz, logdet, &elbo);
  }

  if (tid == 0) out[b] = -elbo / (float)TT;
}

// ---------------- host-side launch ----------------
extern "C" void kernel_launch(void* const* d_in, const int* in_sizes, int n_in,
                              void* d_out, int out_size, void* d_ws, size_t ws_size,
                              hipStream_t stream) {
  (void)in_sizes; (void)n_in; (void)out_size;
  const float* x     = (const float*)d_in[0];
  const float* u     = (const float*)d_in[1];
  const float* Rz    = (const float*)d_in[2];
  const float* Rz0   = (const float*)d_in[3];
  const float* Rx    = (const float*)d_in[4];
  const float* Bobs  = (const float*)d_in[5];
  const float* Obias = (const float*)d_in[6];
  const float* Mw    = (const float*)d_in[7];
  const float* Nw    = (const float*)d_in[8];
  const float* Wu    = (const float*)d_in[9];
  const float* Hb    = (const float*)d_in[10];
  const float* W0    = (const float*)d_in[11];
  const float* b0    = (const float*)d_in[12];
  float* out = (float*)d_out;

  const long long nx = (long long)BB * DX * TT;
  const long long nu = (long long)BB * DU * TT;
  const size_t need = (size_t)(nx + nu) * sizeof(float);

  const float* xbuf = x;
  const float* ubuf = u;
  int tr_layout = 0;
  if (ws_size >= need) {
    float* xT = (float*)d_ws;
    float* uT = xT + nx;
    transpose_bdt<<<(unsigned)((nx + NTHR - 1) / NTHR), NTHR, 0, stream>>>(x, xT, DX, nx);
    transpose_bdt<<<(unsigned)((nu + NTHR - 1) / NTHR), NTHR, 0, stream>>>(u, uT, DU, nu);
    xbuf = xT;
    ubuf = uT;
    tr_layout = 1;
  }

  pf_elbo_kernel<<<BB, NTHR, sizeof(Smem), stream>>>(
      xbuf, ubuf, tr_layout, Rz, Rz0, Rx, Bobs, Obias, Mw, Nw, Wu, Hb, W0, b0, out);
}
